// TransposedAttention_46866683134122
// MI455X (gfx1250) — matrix operand order repaired and checked by reading, not compiled, not run on hardware
//
#include <hip/hip_runtime.h>
#include <hip/hip_bf16.h>

typedef __attribute__((ext_vector_type(16))) _Float16 v16h;
typedef __attribute__((ext_vector_type(8)))  _Float16 v8h;
typedef __attribute__((ext_vector_type(8)))  float    v8f;

// ---------------------------------------------------------------------------
// WMMA fragment loaders per CDNA5 ISA 7.12.2 (16-bit layouts, wave32).
//  A (16x32, row-major src, row stride ld halves):
//    lane holds row (lane&15); K = {kh..kh+7} u {16+kh..16+kh+7}, kh=8*(lane>>4)
//  B (32x16, src laid out [col][k], row stride ld halves):
//    lane holds col (lane&15); K = {kh..kh+7} u {16+kh..16+kh+7}, kh=8*(lane>>4)
//  C/D (16x16 f32): VGPR r -> row m0 + 8*(lane>>4) + r, col n0 + (lane&15)
// ---------------------------------------------------------------------------
__device__ __forceinline__ v16h cat8(v8h lo, v8h hi) {
  return __builtin_shufflevector(lo, hi, 0,1,2,3,4,5,6,7,8,9,10,11,12,13,14,15);
}

__device__ __forceinline__ v16h load_a_frag(const _Float16* __restrict__ base,
                                            int ld, int m0, int k0) {
  int lane = threadIdx.x & 31;
  const _Float16* p = base + (size_t)(m0 + (lane & 15)) * ld + k0 + ((lane >> 4) * 8);
  v8h lo = *(const v8h*)p;
  v8h hi = *(const v8h*)(p + 16);
  return cat8(lo, hi);
}

__device__ __forceinline__ v16h load_b_frag(const _Float16* __restrict__ base,
                                            int ld, int n0, int k0) {
  int lane = threadIdx.x & 31;
  const _Float16* p = base + (size_t)(n0 + (lane & 15)) * ld + k0 + ((lane >> 4) * 8);
  v8h lo = *(const v8h*)p;
  v8h hi = *(const v8h*)(p + 16);
  return cat8(lo, hi);
}

// Problem constants
#define BB   4
#define CC   64
#define NH   4
#define HD   16
#define NPIX 65536           // 256*256
#define C3   192             // 3*C

// ---------------------------------------------------------------------------
// K0: zero accumulators, convert qkv_w (192x64 fp32) -> f16
// ---------------------------------------------------------------------------
__global__ __launch_bounds__(256) void k_init(const float* __restrict__ qkv_w,
                                              _Float16* __restrict__ Wh,
                                              float* __restrict__ G,
                                              float* __restrict__ qn,
                                              float* __restrict__ kn) {
  int i = blockIdx.x * 256 + threadIdx.x;
  if (i < C3 * CC) Wh[i] = (_Float16)qkv_w[i];
  if (i < BB * NH * HD * HD) G[i] = 0.0f;
  if (i < BB * NH * HD) { qn[i] = 0.0f; kn[i] = 0.0f; }
}

// ---------------------------------------------------------------------------
// K1: qkv 1x1 conv as GEMM  tmp1[b][oc][n] = sum_ic W[oc][ic]*x[b][ic][n]+bias
// M=192 (12 m-tiles), K=64 (2 k-steps), 64 pixels per workgroup.
// x tile staged transposed in LDS (f16) so B-fragments are contiguous.
// ---------------------------------------------------------------------------
__global__ __launch_bounds__(256) void k_conv1x1_qkv(const float* __restrict__ x,
                                                     const _Float16* __restrict__ Wh,
                                                     const float* __restrict__ qkv_b,
                                                     _Float16* __restrict__ tmp1) {
  __shared__ _Float16 xT[64 * 72];            // [n][k], padded stride 72
  const int n0 = blockIdx.x * 64;
  const int b  = blockIdx.y;
  const float* xb = x + (size_t)b * CC * NPIX;

  int t  = threadIdx.x;
  int nn = t & 63;
  int kg = t >> 6;                            // 4 groups of 16 k-rows
#pragma unroll
  for (int i = 0; i < 16; ++i) {
    int k = kg * 16 + i;
    xT[nn * 72 + k] = (_Float16)xb[(size_t)k * NPIX + n0 + nn];
  }
  __syncthreads();

  int wave = t >> 5, lane = t & 31;
  _Float16* out = tmp1 + (size_t)b * C3 * NPIX;
#pragma unroll
  for (int it = 0; it < 6; ++it) {            // 8 waves x 6 = 48 tiles = 12m x 4n
    int tid = wave * 6 + it;
    int m = tid >> 2, nt = tid & 3;
    v8f acc = {};
#pragma unroll
    for (int ks = 0; ks < 64; ks += 32) {
      v16h a  = load_a_frag(Wh, CC, m * 16, ks);
      v16h bf = load_b_frag(xT, 72, nt * 16, ks);
      acc = __builtin_amdgcn_wmma_f32_16x16x32_f16(false, a, false, bf,
                                                   (short)0, acc, false, false);
    }
    int col   = n0 + nt * 16 + (lane & 15);
    int rbase = m * 16 + (lane >> 4) * 8;
#pragma unroll
    for (int r = 0; r < 8; ++r) {
      float v = acc[r] + qkv_b[rbase + r];
      out[(size_t)(rbase + r) * NPIX + col] = (_Float16)v;
    }
  }
}

// ---------------------------------------------------------------------------
// K2: depthwise 3x3 + bias, vectorized: 8 contiguous pixels per thread.
// Per thread: 3 aligned v8h row loads + 2 edge halves per row, one v8h store.
// ---------------------------------------------------------------------------
__global__ __launch_bounds__(256) void k_dwconv(const _Float16* __restrict__ tmp1,
                                                const float* __restrict__ dw_w,
                                                const float* __restrict__ dw_b,
                                                _Float16* __restrict__ qkv) {
  size_t tid = (size_t)blockIdx.x * 256 + threadIdx.x;   // over 4*192*65536/8
  int x0 = (int)((tid & 31) << 3);          // 32 threads cover one 256-px row
  int yy = (int)((tid >> 5) & 255);
  size_t cc = tid >> 13;                    // b*192 + c
  int c = (int)(cc % C3);
  const _Float16* plane = tmp1 + (cc << 16);
  const float* w = dw_w + c * 9;
  float bias = dw_b[c];

  float rows[3][10];
#pragma unroll
  for (int r = 0; r < 3; ++r) {
    int y = yy + r - 1;
    bool yin = (unsigned)y < 256u;
    const _Float16* rp = plane + y * 256 + x0;
    v8h mid = {};
    if (yin) mid = *(const v8h*)rp;
#pragma unroll
    for (int j = 0; j < 8; ++j) rows[r][j + 1] = (float)mid[j];
    rows[r][0] = (yin && x0 > 0)       ? (float)rp[-1] : 0.0f;
    rows[r][9] = (yin && x0 + 8 < 256) ? (float)rp[8]  : 0.0f;
  }
  float w00=w[0],w01=w[1],w02=w[2],w10=w[3],w11=w[4],w12=w[5],w20=w[6],w21=w[7],w22=w[8];
  v8h outv;
#pragma unroll
  for (int j = 0; j < 8; ++j) {
    float acc = bias
      + w00*rows[0][j] + w01*rows[0][j+1] + w02*rows[0][j+2]
      + w10*rows[1][j] + w11*rows[1][j+1] + w12*rows[1][j+2]
      + w20*rows[2][j] + w21*rows[2][j+1] + w22*rows[2][j+2];
    outv[j] = (_Float16)acc;
  }
  *(v8h*)(qkv + (cc << 16) + yy * 256 + x0) = outv;
}

// ---------------------------------------------------------------------------
// K3: per (b,h) Gram matrix G = Q K^T plus sum-of-squares norms, WMMA with the
// pixel axis as the reduction (K) dimension. Fragments load straight from
// global (layouts are contiguous 16B/32B runs per lane). atomicAdd f32 accum.
// global_prefetch_b8 issued 4 k-steps (128 pixels) ahead to hide HBM latency.
// ---------------------------------------------------------------------------
__global__ __launch_bounds__(256) void k_gram(const _Float16* __restrict__ qkv,
                                              float* __restrict__ G,
                                              float* __restrict__ qn,
                                              float* __restrict__ kn) {
  int chunk = blockIdx.x;                 // 16 chunks of 4096 pixels
  int bh    = blockIdx.y;                 // b*4 + h
  int b = bh >> 2, h = bh & 3;
  int wave = threadIdx.x >> 5, lane = threadIdx.x & 31;

  const _Float16* qb = qkv + (size_t)b * C3 * NPIX + (size_t)(h * HD) * NPIX;
  const _Float16* kb = qkv + (size_t)b * C3 * NPIX + (size_t)(CC + h * HD) * NPIX;

  int nbase = chunk * 4096 + wave * 512;  // 16 k-steps of 32 per wave
  v8f acc = {};
  float qss = 0.0f, kss = 0.0f;
#pragma unroll 4
  for (int s = 0; s < 512; s += 32) {
    int k0 = nbase + s;
    // prefetch the cachelines this lane will need 4 k-steps from now
    __builtin_prefetch((const void*)(qb + (size_t)(lane & 15) * NPIX + k0 + 128), 0, 1);
    __builtin_prefetch((const void*)(kb + (size_t)(lane & 15) * NPIX + k0 + 128), 0, 1);
    v16h a  = load_a_frag(qb, NPIX, 0, k0);   // rows = q channels of head h
    v16h bf = load_b_frag(kb, NPIX, 0, k0);   // cols = k channels of head h
#pragma unroll
    for (int e = 0; e < 16; ++e) {
      float av = (float)a[e];  qss += av * av;   // all elems belong to row lane&15
      float bv = (float)bf[e]; kss += bv * bv;   // all elems belong to col lane&15
    }
    acc = __builtin_amdgcn_wmma_f32_16x16x32_f16(false, a, false, bf,
                                                 (short)0, acc, false, false);
  }
  float* Gbh = G + bh * (HD * HD);
  int colD  = lane & 15;
  int rbase = (lane >> 4) * 8;
#pragma unroll
  for (int r = 0; r < 8; ++r)
    atomicAdd(&Gbh[(rbase + r) * HD + colD], acc[r]);
  atomicAdd(&qn[bh * HD + (lane & 15)], qss);
  atomicAdd(&kn[bh * HD + colD], kss);
}

// ---------------------------------------------------------------------------
// K4: finalize attn = softmax(G / (||q|| ||k||) * temp) per head, then fold
// proj into a single 64x64 matrix per batch: M = proj_w @ blockdiag(attn).
// ---------------------------------------------------------------------------
__global__ __launch_bounds__(64) void k_attn_mcomb(const float* __restrict__ G,
                                                   const float* __restrict__ qn,
                                                   const float* __restrict__ kn,
                                                   const float* __restrict__ temperature,
                                                   const float* __restrict__ proj_w,
                                                   _Float16* __restrict__ Mcomb) {
  __shared__ float attn[CC][HD];          // [h*16+c][d]
  int b = blockIdx.x;
  int t = threadIdx.x;                    // 0..63 -> (h,c)
  int h = t >> 4, c = t & 15;
  int bh = b * NH + h;
  const float* Gr = G + bh * (HD * HD) + c * HD;
  float qd   = fmaxf(sqrtf(qn[bh * HD + c]), 1e-12f);
  float temp = temperature[h];
  float row[HD], mx = -1e30f;
#pragma unroll
  for (int d = 0; d < HD; ++d) {
    float kd = fmaxf(sqrtf(kn[bh * HD + d]), 1e-12f);
    row[d] = Gr[d] / (qd * kd) * temp;
    mx = fmaxf(mx, row[d]);
  }
  float s = 0.0f;
#pragma unroll
  for (int d = 0; d < HD; ++d) { row[d] = __expf(row[d] - mx); s += row[d]; }
  float inv = 1.0f / s;
#pragma unroll
  for (int d = 0; d < HD; ++d) attn[t][d] = row[d] * inv;
  __syncthreads();

  int o = t;                              // output channel
  for (int dg = 0; dg < CC; ++dg) {
    int hh = dg >> 4, dd = dg & 15;
    float acc = 0.0f;
#pragma unroll
    for (int c2 = 0; c2 < HD; ++c2)
      acc += proj_w[o * CC + hh * HD + c2] * attn[hh * HD + c2][dd];
    Mcomb[(b * CC + o) * CC + dg] = (_Float16)acc;
  }
}

// ---------------------------------------------------------------------------
// K5: y = Mcomb @ V + proj_b   (M=64, K=64, 128 pixels per workgroup)
// Each wave owns one 16-pixel column and sweeps all 4 m-tiles -> B frag reuse.
// ---------------------------------------------------------------------------
__global__ __launch_bounds__(256) void k_out_gemm(const _Float16* __restrict__ qkv,
                                                  const _Float16* __restrict__ Mcomb,
                                                  const float* __restrict__ proj_b,
                                                  float* __restrict__ out) {
  __shared__ _Float16 vT[128 * 72];       // [n][k] padded
  const int n0 = blockIdx.x * 128;
  const int b  = blockIdx.y;
  const _Float16* vb = qkv + (size_t)b * C3 * NPIX + (size_t)(2 * CC) * NPIX;

  int t  = threadIdx.x;
  int nn = t & 127;
  int kg = t >> 7;                         // 2 groups of 32 k-rows
#pragma unroll
  for (int i = 0; i < 32; ++i) {
    int k = kg * 32 + i;
    vT[nn * 72 + k] = vb[(size_t)k * NPIX + n0 + nn];
  }
  __syncthreads();

  const _Float16* A = Mcomb + (size_t)b * CC * CC;
  int wave = t >> 5, lane = t & 31;
  float* ob = out + (size_t)b * CC * NPIX;
#pragma unroll
  for (int m = 0; m < 4; ++m) {
    v8f acc = {};
#pragma unroll
    for (int ks = 0; ks < 64; ks += 32) {
      v16h a  = load_a_frag(A, CC, m * 16, ks);
      v16h bf = load_b_frag(vT, 72, wave * 16, ks);
      acc = __builtin_amdgcn_wmma_f32_16x16x32_f16(false, a, false, bf,
                                                   (short)0, acc, false, false);
    }
    int col   = n0 + wave * 16 + (lane & 15);
    int rbase = m * 16 + (lane >> 4) * 8;
#pragma unroll
    for (int r = 0; r < 8; ++r)
      ob[(size_t)(rbase + r) * NPIX + col] = acc[r] + proj_b[rbase + r];
  }
}

// ---------------------------------------------------------------------------
extern "C" void kernel_launch(void* const* d_in, const int* in_sizes, int n_in,
                              void* d_out, int out_size, void* d_ws, size_t ws_size,
                              hipStream_t stream) {
  const float* x      = (const float*)d_in[0];
  const float* qkv_w  = (const float*)d_in[1];
  const float* qkv_b  = (const float*)d_in[2];
  const float* dw_w   = (const float*)d_in[3];
  const float* dw_b   = (const float*)d_in[4];
  const float* proj_w = (const float*)d_in[5];
  const float* proj_b = (const float*)d_in[6];
  const float* temp   = (const float*)d_in[7];

  char* ws = (char*)d_ws;
  const size_t QKV_BYTES = (size_t)BB * C3 * NPIX * sizeof(_Float16); // 96 MB
  _Float16* tmp1  = (_Float16*)(ws);
  _Float16* qkv   = (_Float16*)(ws + QKV_BYTES);
  char*     small = ws + 2 * QKV_BYTES;
  _Float16* Wh    = (_Float16*)small;                 // 192*64 f16
  float*    G     = (float*)(small + 32768);          // 4*4*16*16 f32
  float*    qn    = (float*)(small + 32768 + 16384);  // 256 f32
  float*    kn    = qn + 256;
  _Float16* Mcomb = (_Float16*)(small + 32768 + 16384 + 4096); // 4*64*64 f16

  // K0: init accumulators + f16 weights
  k_init<<<48, 256, 0, stream>>>(qkv_w, Wh, G, qn, kn);

  // K1: qkv 1x1 conv (WMMA GEMM, f16 out)
  dim3 g1(NPIX / 64, BB);
  k_conv1x1_qkv<<<g1, 256, 0, stream>>>(x, Wh, qkv_b, tmp1);

  // K2: depthwise 3x3 (8 px/thread, vectorized)
  k_dwconv<<<(BB * C3 * NPIX / 8) / 256, 256, 0, stream>>>(tmp1, dw_w, dw_b, qkv);

  // K3: Gram + norms (WMMA reduction over pixels)
  dim3 g3(16, BB * NH);
  k_gram<<<g3, 256, 0, stream>>>(qkv, G, qn, kn);

  // K4: softmax + fold proj into Mcomb
  k_attn_mcomb<<<BB, 64, 0, stream>>>(G, qn, kn, temp, proj_w, Mcomb);

  // K5: y = Mcomb @ V + proj_b (WMMA GEMM, fp32 out)
  dim3 g5(NPIX / 128, BB);
  k_out_gemm<<<g5, 256, 0, stream>>>(qkv, Mcomb, proj_b, (float*)d_out);
}
